// VectorQuantizer_21990232555697
// MI455X (gfx1250) — matrix operand order repaired and checked by reading, not compiled, not run on hardware
//
#include <hip/hip_runtime.h>

#define D_MODEL 512
#define KCODES 8192
#define NROWS 32768
#define STAGE_CODES 64
#define NSTAGES (KCODES / STAGE_CODES)       // 128
#define PSTRIDE 520                          // padded LDS row stride in bf16 elems (avoids bank conflicts)
#define LOSS_SCALE (1.25f / 16777216.0f)     // 1.25 / (NROWS * D_MODEL)

typedef __attribute__((ext_vector_type(16))) __bf16 v16bf;
typedef __attribute__((ext_vector_type(8)))  float  v8f;

union BVec { uint4 u[2]; v16bf v; };

static __device__ __forceinline__ v8f wmma_bf16(v16bf a, v16bf b, v8f c) {
  // (neg_a, A, neg_b, B, c_mod, C, reuse_a, reuse_b)
  return __builtin_amdgcn_wmma_f32_16x16x32_bf16(false, a, false, b, (short)0, c, false, false);
}

// Async DMA of one 64-code stage (64 KB) of bf16 emb into (padded) LDS.
// Per wave: 16 x global_load_async_to_lds_b128 (32 lanes x 16 B each), ASYNCcnt-tracked.
static __device__ __forceinline__ void async_stage_copy(const unsigned short* __restrict__ gsrc,
                                                        unsigned short* sdst, int tid) {
#pragma unroll
  for (int t = 0; t < 16; ++t) {
    const int i = tid + t * 256;
    const int r = i >> 6, sgm = i & 63;
    const unsigned lds_addr = (unsigned)(uintptr_t)(sdst + r * PSTRIDE + sgm * 8);
    const unsigned goff = (unsigned)(i * 16);
    asm volatile("global_load_async_to_lds_b128 %0, %1, %2 offset:0"
                 :: "v"(lds_addr), "v"(goff), "s"(gsrc)
                 : "memory");
  }
}

// ---------------------------------------------------------------------------
// Kernel 1: emb f32 -> bf16 (ws) + per-code squared norms; zero the loss cell.
// ---------------------------------------------------------------------------
__global__ __launch_bounds__(256) void vq_prep(const float* __restrict__ emb,
                                               unsigned short* __restrict__ embh,
                                               float* __restrict__ enorm,
                                               float* __restrict__ loss) {
  if (blockIdx.x == 0 && threadIdx.x == 0) *loss = 0.0f;
  const int row  = blockIdx.x * 8 + (threadIdx.x >> 5);
  const int lane = threadIdx.x & 31;
  const float* e = emb + (size_t)row * D_MODEL;
  unsigned short* o = embh + (size_t)row * D_MODEL;
  float nrm = 0.f;
#pragma unroll
  for (int t = 0; t < 16; ++t) {
    float f = e[lane + t * 32];
    nrm += f * f;
    o[lane + t * 32] = __builtin_bit_cast(unsigned short, (__bf16)f);
  }
#pragma unroll
  for (int off = 16; off > 0; off >>= 1) nrm += __shfl_down(nrm, off, 32);
  if (lane == 0) enorm[row] = nrm;
}

// ---------------------------------------------------------------------------
// Kernel 2: WMMA distance argmin.
// 8 waves/WG; each wave owns 16 rows (A resident in VGPRs, full K=512).
// emb streamed through double-buffered LDS via async-to-LDS DMA, overlapped
// with 4 interleaved WMMA accumulator chains + B-fragment double buffering.
// ---------------------------------------------------------------------------
__global__ __launch_bounds__(256) void vq_argmin(const float* __restrict__ z,
                                                 const unsigned short* __restrict__ embh,
                                                 const float* __restrict__ enorm,
                                                 float* __restrict__ out_idx) {
  __shared__ unsigned short smem[2][STAGE_CODES * PSTRIDE];   // 2 x ~65 KB

  const int tid  = threadIdx.x;
  const int lane = tid & 31;
  const int wid  = tid >> 5;          // wave id 0..7
  const int m    = lane & 15;
  const int h    = lane >> 4;         // half-wave select
  const int rowbase = blockIdx.x * 128 + wid * 16;

  // --- Preload A fragments: 16 rows x K=512 in ISA 16-bit A layout ---
  // lane (m,h): elements 0..7  <-> K = kt*32 + h*8  + {0..7}
  //             elements 8..15 <-> K = kt*32 + 16 + h*8 + {0..7}
  v16bf afrag[16];
  {
    const float* zr = z + (size_t)(rowbase + m) * D_MODEL;
#pragma unroll
    for (int kt = 0; kt < 16; ++kt) {
      const float4 p0 = *(const float4*)(zr + kt * 32 + h * 8);
      const float4 p1 = *(const float4*)(zr + kt * 32 + h * 8 + 4);
      const float4 p2 = *(const float4*)(zr + kt * 32 + 16 + h * 8);
      const float4 p3 = *(const float4*)(zr + kt * 32 + 16 + h * 8 + 4);
      v16bf a;
      a[0]  = (__bf16)p0.x; a[1]  = (__bf16)p0.y; a[2]  = (__bf16)p0.z; a[3]  = (__bf16)p0.w;
      a[4]  = (__bf16)p1.x; a[5]  = (__bf16)p1.y; a[6]  = (__bf16)p1.z; a[7]  = (__bf16)p1.w;
      a[8]  = (__bf16)p2.x; a[9]  = (__bf16)p2.y; a[10] = (__bf16)p2.z; a[11] = (__bf16)p2.w;
      a[12] = (__bf16)p3.x; a[13] = (__bf16)p3.y; a[14] = (__bf16)p3.z; a[15] = (__bf16)p3.w;
      afrag[kt] = a;
    }
  }

  float minv[8];
  int   mini[8];
#pragma unroll
  for (int j = 0; j < 8; ++j) { minv[j] = 3.402823466e38f; mini[j] = 0; }

  // B-fragment loader: 4 code-tiles at k-chunk kt.
  // B layout (32x16, 16-bit): lane n = column, elements 0..7 <-> K = h*8 + {0..7}, 8..15 <-> K = 16 + h*8 + {0..7}
  auto loadB = [&](BVec* bv, const unsigned short* sbuf, int kt) {
#pragma unroll
    for (int t = 0; t < 4; ++t) {
      const unsigned short* p = sbuf + (t * 16 + m) * PSTRIDE + kt * 32 + h * 8;
      bv[t].u[0] = *(const uint4*)p;
      bv[t].u[1] = *(const uint4*)(p + 16);
    }
  };

  // Prologue: DMA stage 0 into buffer 0.
  async_stage_copy(embh, &smem[0][0], tid);
  asm volatile("s_wait_asynccnt 0x0" ::: "memory");
  __syncthreads();

  for (int stage = 0; stage < NSTAGES; ++stage) {
    unsigned short* sbuf = &smem[stage & 1][0];

    // Kick off DMA of the next stage into the other buffer; overlaps the WMMAs below.
    if (stage + 1 < NSTAGES)
      async_stage_copy(embh + (size_t)(stage + 1) * STAGE_CODES * D_MODEL,
                       &smem[(stage + 1) & 1][0], tid);

    v8f acc0 = {0,0,0,0,0,0,0,0};
    v8f acc1 = {0,0,0,0,0,0,0,0};
    v8f acc2 = {0,0,0,0,0,0,0,0};
    v8f acc3 = {0,0,0,0,0,0,0,0};

    BVec cur[4], nxt[4];
    loadB(cur, sbuf, 0);
#pragma unroll
    for (int kt = 0; kt < 16; ++kt) {
      if (kt < 15) loadB(nxt, sbuf, kt + 1);       // issue next B loads ahead of the WMMAs
      acc0 = wmma_bf16(afrag[kt], cur[0].v, acc0);
      acc1 = wmma_bf16(afrag[kt], cur[1].v, acc1);
      acc2 = wmma_bf16(afrag[kt], cur[2].v, acc2);
      acc3 = wmma_bf16(afrag[kt], cur[3].v, acc3);
      if (kt < 15) { cur[0] = nxt[0]; cur[1] = nxt[1]; cur[2] = nxt[2]; cur[3] = nxt[3]; }
    }

    // score = ||e||^2 - 2 * (x . e) ; running min per row (C: VGPR j = row h*8+j, lane = col)
    const int cb = stage * STAGE_CODES + m;
    const float en0 = enorm[cb];
    const float en1 = enorm[cb + 16];
    const float en2 = enorm[cb + 32];
    const float en3 = enorm[cb + 48];
#pragma unroll
    for (int j = 0; j < 8; ++j) {
      float s;
      s = en0 - 2.0f * acc0[j]; if (s < minv[j]) { minv[j] = s; mini[j] = cb;      }
      s = en1 - 2.0f * acc1[j]; if (s < minv[j]) { minv[j] = s; mini[j] = cb + 16; }
      s = en2 - 2.0f * acc2[j]; if (s < minv[j]) { minv[j] = s; mini[j] = cb + 32; }
      s = en3 - 2.0f * acc3[j]; if (s < minv[j]) { minv[j] = s; mini[j] = cb + 48; }
    }

    // My async DMAs done; barrier makes everyone's DMA + my LDS reads globally ordered.
    asm volatile("s_wait_asynccnt 0x0" ::: "memory");
    __syncthreads();
  }

  // Cross-lane min across the 16 columns (stay within each 16-lane half).
#pragma unroll
  for (int j = 0; j < 8; ++j) {
    float v = minv[j];
    int   bi = mini[j];
#pragma unroll
    for (int mask = 8; mask > 0; mask >>= 1) {
      float ov = __shfl_xor(v, mask, 32);
      int   oi = __shfl_xor(bi, mask, 32);
      if (ov < v || (ov == v && oi < bi)) { v = ov; bi = oi; }
    }
    if (m == 0) out_idx[rowbase + h * 8 + j] = (float)bi;   // lanes 0 and 16 write
  }
}

// ---------------------------------------------------------------------------
// Kernel 3: gather emb[idx] (f32), write straight-through output, reduce loss.
// ---------------------------------------------------------------------------
__global__ __launch_bounds__(256) void vq_gather(const float* __restrict__ z,
                                                 const float* __restrict__ emb,
                                                 const float* __restrict__ idxf,
                                                 float* __restrict__ out_q,
                                                 float* __restrict__ loss) {
  const int row = blockIdx.x;
  const int tid = threadIdx.x;
  const int idx = (int)idxf[row];
  const float* e = emb + (size_t)idx * D_MODEL;
  const float* x = z + (size_t)row * D_MODEL;
  float* q = out_q + (size_t)row * D_MODEL;
  float sse = 0.f;
#pragma unroll
  for (int t = 0; t < 2; ++t) {
    const int c = tid + t * 256;
    const float qv = e[c], xv = x[c];
    const float d = qv - xv;
    sse += d * d;
    q[c] = xv + d;                 // flat + (quantized - flat), matches reference arithmetic
  }
#pragma unroll
  for (int off = 16; off > 0; off >>= 1) sse += __shfl_down(sse, off, 32);
  __shared__ float wsum[8];
  if ((tid & 31) == 0) wsum[tid >> 5] = sse;
  __syncthreads();
  if (tid == 0) {
    float t = 0.f;
#pragma unroll
    for (int i = 0; i < 8; ++i) t += wsum[i];
    atomicAdd(loss, t * LOSS_SCALE);
  }
}

// ---------------------------------------------------------------------------
extern "C" void kernel_launch(void* const* d_in, const int* in_sizes, int n_in,
                              void* d_out, int out_size, void* d_ws, size_t ws_size,
                              hipStream_t stream) {
  const float* z   = (const float*)d_in[0];   // [32,1024,512] f32
  const float* emb = (const float*)d_in[1];   // [8192,512]   f32

  float* out      = (float*)d_out;
  float* out_loss = out;                                   // [1]
  float* out_q    = out + 1;                               // [NROWS*D]
  float* out_idx  = out + 1 + (size_t)NROWS * D_MODEL;     // [NROWS]

  unsigned short* embh = (unsigned short*)d_ws;                                        // 8 MB bf16
  float* enorm = (float*)((char*)d_ws + (size_t)KCODES * D_MODEL * sizeof(unsigned short)); // 32 KB

  vq_prep  <<<KCODES / 8,  256, 0, stream>>>(emb, embh, enorm, out_loss);
  vq_argmin<<<NROWS / 128, 256, 0, stream>>>(z, embh, enorm, out_idx);
  vq_gather<<<NROWS,       256, 0, stream>>>(z, emb, out_idx, out_q, out_loss);
}
